// GatNet_9474697855739
// MI455X (gfx1250) — matrix operand order repaired and checked by reading, not compiled, not run on hardware
//
#include <hip/hip_runtime.h>
#include <hip/hip_bf16.h>
#include <math.h>

// ---------------------------------------------------------------------------
// TransformerConv GNN layer for gfx1250 (MI455X).
// GEMMs: bf16 WMMA (v_wmma_f32_16x16x32_bf16) with the per-block weight slice
// staged into LDS via the Tensor Data Mover (tensor_load_to_lds + TENSORcnt).
// Edge phase: L2-resident gather/scatter with float atomics.
// ---------------------------------------------------------------------------

typedef __bf16 bf16_t;
typedef __attribute__((ext_vector_type(16))) __bf16 v16bf;
typedef __attribute__((ext_vector_type(8)))  __bf16 v8bf;
typedef __attribute__((ext_vector_type(8)))  float  v8f;
typedef __attribute__((ext_vector_type(4)))  unsigned int u32x4;
typedef __attribute__((ext_vector_type(8)))  int i32x8;
typedef __attribute__((ext_vector_type(4)))  int i32x4;

#define F_DIM 128
#define K_DIM 128
#define COA   640            // [q|k|v|skip|gate_top]
#define COB   256            // [gate_bot|fuse]
#define NKT   (K_DIM / 32)   // 4 k-tiles of 32
#define WA_ELEMS ((COA / 16) * NKT * 512)   // packed B fragments, 16 bf16/lane
#define WB_ELEMS ((COB / 16) * NKT * 512)
#define B_SLICE_ELEMS (4 * NKT * 512)       // 8192 bf16 = 16 KB per block tile
#define B_SLICE_DW    (B_SLICE_ELEMS / 2)   // 4096 dwords

__device__ __forceinline__ bf16_t f2bf(float f) {
  // round-to-nearest-even truncation f32 -> bf16 (bit-level, no cvt dependency)
  unsigned u = __float_as_uint(f);
  u += 0x7FFFu + ((u >> 16) & 1u);
  unsigned short s = (unsigned short)(u >> 16);
  bf16_t r;
  __builtin_memcpy(&r, &s, 2);
  return r;
}

// Monotone encode/decode of float into uint for atomicMax-based segment max.
__device__ __forceinline__ unsigned enc_f32(float f) {
  unsigned u = __float_as_uint(f);
  return (u & 0x80000000u) ? ~u : (u | 0x80000000u);
}
__device__ __forceinline__ float dec_f32(unsigned k) {
  return (k & 0x80000000u) ? __uint_as_float(k ^ 0x80000000u)
                           : __uint_as_float(~k);
}

// ---------------------------------------------------------------------------
// Init: agg = 0, denom = 0, amax = enc(-inf)
// ---------------------------------------------------------------------------
__global__ void k_init(float* __restrict__ agg, float* __restrict__ denom,
                       unsigned* __restrict__ amax, int N) {
  int tid = blockIdx.x * blockDim.x + threadIdx.x;
  if (tid < N * F_DIM) agg[tid] = 0.0f;
  if (tid < N * 2) {
    denom[tid] = 0.0f;
    amax[tid] = 0x007FFFFFu;   // enc(-inf)
  }
}

// ---------------------------------------------------------------------------
// Pack weights into WMMA B-fragment layout (bf16) + fused bias vector.
// Packed index: (((ntile*NKT + kt)*32 + lane)*16 + i)
// B lane layout (16-bit, K=32): n = ntile*16 + lane%16,
//                               k = kt*32 + (i>=8 ? 16 : 0) + (lane>=16 ? 8 : 0) + (i&7)
// ---------------------------------------------------------------------------
__global__ void k_pack(const float* __restrict__ Wq, const float* __restrict__ Wk,
                       const float* __restrict__ Wv, const float* __restrict__ Wskip,
                       const float* __restrict__ Wgate, const float* __restrict__ Wfuse,
                       const float* __restrict__ bq, const float* __restrict__ bk,
                       const float* __restrict__ bv, const float* __restrict__ bskip,
                       bf16_t* __restrict__ WAp, bf16_t* __restrict__ WBp,
                       float* __restrict__ biasA) {
  int tid = blockIdx.x * blockDim.x + threadIdx.x;
  if (tid < WA_ELEMS + WB_ELEMS) {
    bool isA = tid < WA_ELEMS;
    int p = isA ? tid : tid - WA_ELEMS;
    int i = p & 15;
    int rest = p >> 4;
    int lane = rest & 31;
    int rest2 = rest >> 5;
    int kt = rest2 & (NKT - 1);
    int ntile = rest2 >> 2;
    int k = kt * 32 + ((i & 8) << 1) + ((lane >= 16) ? 8 : 0) + (i & 7);
    int n = ntile * 16 + (lane & 15);
    float val;
    if (isA) {
      // columns: 0-127 q, 128-255 k, 256-383 v, 384-511 skip, 512-639 gate_top
      if      (n < 128) val = Wq[k * F_DIM + n];
      else if (n < 256) val = Wk[k * F_DIM + (n - 128)];
      else if (n < 384) val = Wv[k * F_DIM + (n - 256)];
      else if (n < 512) val = Wskip[k * F_DIM + (n - 384)];
      else              val = Wgate[k * F_DIM + (n - 512)];          // top half rows
      WAp[p] = f2bf(val);
    } else {
      // columns: 0-127 gate_bot (Wgate rows 128..255), 128-255 fuse
      if (n < 128) val = Wgate[(128 + k) * F_DIM + n];
      else         val = Wfuse[k * F_DIM + (n - 128)];
      WBp[p] = f2bf(val);
    }
  } else if (tid < WA_ELEMS + WB_ELEMS + COA) {
    int c = tid - (WA_ELEMS + WB_ELEMS);
    float b;
    if      (c < 128) b = bq[c];
    else if (c < 256) b = bk[c - 128];
    else if (c < 384) b = bv[c - 256];
    else if (c < 512) b = bskip[c - 384];
    else              b = 0.0f;
    biasA[c] = b;
  }
}

__global__ void k_f32_to_bf16(const float* __restrict__ x, bf16_t* __restrict__ y,
                              int n) {
  int tid = blockIdx.x * blockDim.x + threadIdx.x;
  if (tid < n) y[tid] = f2bf(x[tid]);
}

// ---------------------------------------------------------------------------
// WMMA GEMM: C[M x Co] = A[M x K, bf16 row-major] @ Bp(packed) + bias
// block = 256 threads = 8 waves; wave w -> 16-row strip, 4 x 16x16 N tiles.
// Block tile = 128(M) x 64(N).  The block's 16 KB packed-B slice is staged
// into LDS once via the Tensor Data Mover; waves then read fragments with
// ds_load while global loads carry only A rows and C stores.
// A lane layout (16-bit, 16x32): m = lane%16,
//   elems 0..7  -> K = k0 + (lane>=16?8:0) + 0..7
//   elems 8..15 -> K = k0 + 16 + (lane>=16?8:0) + 0..7
// ---------------------------------------------------------------------------
__global__ void k_gemm_bf16_wmma(const bf16_t* __restrict__ A,
                                 const bf16_t* __restrict__ Bp,
                                 const float* __restrict__ bias,
                                 float* __restrict__ C,
                                 int M, int Co) {
  __shared__ bf16_t bsh[B_SLICE_ELEMS];   // 16 KB

  // --- TDM: one tensor_load_to_lds per block for the contiguous B slice ---
  if (threadIdx.x == 0) {
    size_t gaddr = (size_t)(const void*)(Bp + (size_t)blockIdx.y * B_SLICE_ELEMS);
    unsigned lds = (unsigned)(size_t)(void*)bsh;
    // D# group 0: count=1 (valid user descriptor), lds_addr, global_addr, type=2
    u32x4 g0;
    g0[0] = 1u;
    g0[1] = lds;
    g0[2] = (unsigned)(gaddr & 0xFFFFFFFFu);
    g0[3] = (unsigned)((gaddr >> 32) & 0x01FFFFFFu) | 0x80000000u;  // type=2
    // D# group 1: data_size=4B; 1 row of B_SLICE_DW dwords; stride = row len
    i32x8 g1;
    g1[0] = 0x00020000;                                   // data_size=2 (4B)
    g1[1] = (int)((unsigned)B_SLICE_DW << 16);            // tensor_dim0[15:0]
    g1[2] = (int)(((unsigned)B_SLICE_DW >> 16) | (1u << 16)); // td0 hi | td1 lo
    g1[3] = (int)((unsigned)B_SLICE_DW << 16);            // td1 hi(0) | tile_dim0
    g1[4] = 1;                                            // tile_dim1=1
    g1[5] = B_SLICE_DW;                                   // tensor_dim0_stride lo
    g1[6] = (int)((unsigned)B_SLICE_DW << 16);            // s0 hi(0) | s1 lo
    g1[7] = 0;
    i32x4 z4 = {0, 0, 0, 0};
#if defined(__clang_major__) && (__clang_major__ >= 23)
    i32x8 z8 = {0, 0, 0, 0, 0, 0, 0, 0};
    __builtin_amdgcn_tensor_load_to_lds(g0, g1, z4, z4, z8, 0);
#else
    __builtin_amdgcn_tensor_load_to_lds(g0, g1, z4, z4, 0);
#endif
    __builtin_amdgcn_s_wait_tensorcnt(0);
  }
  __syncthreads();

  const int lane = threadIdx.x & 31;
  const int wave = threadIdx.x >> 5;
  const int rowBase = blockIdx.x * 128 + wave * 16;
  const int nBase = blockIdx.y * 64;

  const int m = lane & 15;
  const int halfsel = (lane >> 4) * 8;

  int row = rowBase + m;
  int rowc = row < M ? row : M - 1;
  const bf16_t* arow = A + (size_t)rowc * K_DIM;

  v8f acc[4] = {};

  #pragma unroll
  for (int kt = 0; kt < NKT; ++kt) {
    const int k0 = kt * 32;
    v8bf alo = *(const v8bf*)(arow + k0 + halfsel);
    v8bf ahi = *(const v8bf*)(arow + k0 + 16 + halfsel);
    v16bf afrag;
    #pragma unroll
    for (int i = 0; i < 8; ++i) { afrag[i] = alo[i]; afrag[8 + i] = ahi[i]; }
    #pragma unroll
    for (int t = 0; t < 4; ++t) {
      const bf16_t* bptr = bsh + ((t * NKT + kt) * 512 + lane * 16);
      v16bf bfrag = *(const v16bf*)bptr;
      acc[t] = __builtin_amdgcn_wmma_f32_16x16x32_bf16(
          false, afrag, false, bfrag, (short)0, acc[t], false, false);
    }
  }

  // C/D layout: VGPR j -> row (lane<16 ? j : 8+j), col = lane%16
  const int coloff = lane & 15;
  const int rtop = rowBase + ((lane >> 4) ? 8 : 0);
  #pragma unroll
  for (int t = 0; t < 4; ++t) {
    int col = nBase + t * 16 + coloff;
    float bv = bias ? bias[col] : 0.0f;
    #pragma unroll
    for (int j = 0; j < 8; ++j) {
      int r = rtop + j;
      if (r < M) C[(size_t)r * Co + col] = acc[t][j] + bv;
    }
  }
}

// ---------------------------------------------------------------------------
// Edge pass 1: alpha[e,h] = dot(q[dst], k[src]) / 8 ; segment max via atomicMax
// ---------------------------------------------------------------------------
__global__ void k_edge_logits(const long long* __restrict__ srcp,
                              const long long* __restrict__ dstp,
                              const float* __restrict__ Y1,
                              float* __restrict__ alpha,
                              unsigned* __restrict__ amax, int E) {
  int e = blockIdx.x * blockDim.x + threadIdx.x;
  if (e >= E) return;
  long long s = srcp[e], d = dstp[e];
  const float4* q4 = (const float4*)(Y1 + (size_t)d * COA);        // q: cols 0..127
  const float4* k4 = (const float4*)(Y1 + (size_t)s * COA + 128);  // k: cols 128..255
  float a0 = 0.0f, a1 = 0.0f;
  #pragma unroll
  for (int i = 0; i < 16; ++i) {
    float4 qa = q4[i], ka = k4[i];
    a0 += qa.x * ka.x + qa.y * ka.y + qa.z * ka.z + qa.w * ka.w;
  }
  #pragma unroll
  for (int i = 16; i < 32; ++i) {
    float4 qa = q4[i], ka = k4[i];
    a1 += qa.x * ka.x + qa.y * ka.y + qa.z * ka.z + qa.w * ka.w;
  }
  a0 *= 0.125f;  // 1/sqrt(64)
  a1 *= 0.125f;
  alpha[(size_t)e * 2 + 0] = a0;
  alpha[(size_t)e * 2 + 1] = a1;
  atomicMax(&amax[d * 2 + 0], enc_f32(a0));
  atomicMax(&amax[d * 2 + 1], enc_f32(a1));
}

// ---------------------------------------------------------------------------
// Edge pass 2: ex = exp(alpha - amax[dst]); denom[dst] += ex (in-place on alpha)
// ---------------------------------------------------------------------------
__global__ void k_edge_exp(const long long* __restrict__ dstp,
                           float* __restrict__ alpha,
                           const unsigned* __restrict__ amax,
                           float* __restrict__ denom, int E) {
  int tid = blockIdx.x * blockDim.x + threadIdx.x;
  if (tid >= E * 2) return;
  int e = tid >> 1, h = tid & 1;
  long long d = dstp[e];
  float ex = __expf(alpha[tid] - dec_f32(amax[d * 2 + h]));
  alpha[tid] = ex;
  atomicAdd(&denom[d * 2 + h], ex);
}

// ---------------------------------------------------------------------------
// Edge pass 3: agg[dst] += v[src] * (ex / denom[dst]).  One wave per edge,
// each lane owns 4 features (float4 gather + 4 global_atomic_add_f32).
// ---------------------------------------------------------------------------
__global__ void k_edge_agg(const long long* __restrict__ srcp,
                           const long long* __restrict__ dstp,
                           const float* __restrict__ Y1,
                           const float* __restrict__ ex,
                           const float* __restrict__ denom,
                           float* __restrict__ agg, int E) {
  long long tid = (long long)blockIdx.x * blockDim.x + threadIdx.x;
  int e = (int)(tid >> 5);
  if (e >= E) return;
  int lane = (int)(tid & 31);
  long long s = srcp[e], d = dstp[e];
  int h = lane >> 4;  // lanes 0-15: features 0..63 (head 0); 16-31: head 1
  float w = ex[(size_t)e * 2 + h] / denom[d * 2 + h];
  int f = lane * 4;
  float4 vv = *(const float4*)(Y1 + (size_t)s * COA + 256 + f);  // v: cols 256..383
  float* ag = agg + (size_t)d * F_DIM + f;
  atomicAdd(ag + 0, vv.x * w);
  atomicAdd(ag + 1, vv.y * w);
  atomicAdd(ag + 2, vv.z * w);
  atomicAdd(ag + 3, vv.w * w);
}

// ---------------------------------------------------------------------------
// x = agg + skip ; emit f32 and bf16 copies
// ---------------------------------------------------------------------------
__global__ void k_make_x(const float* __restrict__ agg, const float* __restrict__ Y1,
                         float* __restrict__ xf, bf16_t* __restrict__ xb, int N) {
  int tid = blockIdx.x * blockDim.x + threadIdx.x;
  if (tid >= N * F_DIM) return;
  int n = tid >> 7, f = tid & 127;
  float x = agg[tid] + Y1[(size_t)n * COA + 384 + f];  // skip: cols 384..511
  xf[tid] = x;
  xb[tid] = f2bf(x);
}

// ---------------------------------------------------------------------------
// out = gate * tanh(fuse) + (1-gate) * nodes,  gate = sigmoid(gtop + gbot)
// ---------------------------------------------------------------------------
__global__ void k_output(const float* __restrict__ Y1, const float* __restrict__ Y2,
                         const float* __restrict__ nodes, float* __restrict__ out,
                         int N) {
  int tid = blockIdx.x * blockDim.x + threadIdx.x;
  if (tid >= N * F_DIM) return;
  int n = tid >> 7, f = tid & 127;
  float gl = Y1[(size_t)n * COA + 512 + f] + Y2[(size_t)n * COB + f];
  float g = 1.0f / (1.0f + __expf(-gl));
  float fu = tanhf(Y2[(size_t)n * COB + 128 + f]);
  out[tid] = g * fu + (1.0f - g) * nodes[tid];
}

// ---------------------------------------------------------------------------
static inline size_t align256(size_t x) { return (x + 255) & ~(size_t)255; }

extern "C" void kernel_launch(void* const* d_in, const int* in_sizes, int n_in,
                              void* d_out, int out_size, void* d_ws, size_t ws_size,
                              hipStream_t stream) {
  const float* nodes      = (const float*)d_in[0];
  const long long* ei     = (const long long*)d_in[1];
  const float* Wq         = (const float*)d_in[2];
  const float* bq         = (const float*)d_in[3];
  const float* Wk         = (const float*)d_in[4];
  const float* bk         = (const float*)d_in[5];
  const float* Wv         = (const float*)d_in[6];
  const float* bv         = (const float*)d_in[7];
  const float* Wskip      = (const float*)d_in[8];
  const float* bskip      = (const float*)d_in[9];
  const float* Wgate      = (const float*)d_in[10];
  const float* Wfuse      = (const float*)d_in[11];
  float* out              = (float*)d_out;

  const int N = in_sizes[0] / F_DIM;
  const int E = in_sizes[1] / 2;
  const long long* srcp = ei;
  const long long* dstp = ei + E;

  // workspace carve-up
  char* w = (char*)d_ws;
  size_t off = 0;
  auto take = [&](size_t bytes) -> void* {
    void* p = (void*)(w + off);
    off = align256(off + bytes);
    return p;
  };
  bf16_t*  nodes_bf = (bf16_t*)take((size_t)N * F_DIM * sizeof(bf16_t));
  bf16_t*  x_bf     = (bf16_t*)take((size_t)N * F_DIM * sizeof(bf16_t));
  bf16_t*  WAp      = (bf16_t*)take((size_t)WA_ELEMS * sizeof(bf16_t));
  bf16_t*  WBp      = (bf16_t*)take((size_t)WB_ELEMS * sizeof(bf16_t));
  float*   biasA    = (float*)take((size_t)COA * sizeof(float));
  float*   Y1       = (float*)take((size_t)N * COA * sizeof(float));
  float*   Y2       = (float*)take((size_t)N * COB * sizeof(float));
  float*   alpha    = (float*)take((size_t)E * 2 * sizeof(float));
  unsigned* amax    = (unsigned*)take((size_t)N * 2 * sizeof(unsigned));
  float*   denom    = (float*)take((size_t)N * 2 * sizeof(float));
  float*   agg      = (float*)take((size_t)N * F_DIM * sizeof(float));
  float*   xf       = (float*)take((size_t)N * F_DIM * sizeof(float));
  (void)ws_size; (void)n_in; (void)out_size;

  const int BT = 256;
  const int nNF = N * F_DIM;

  // 1. init accumulators
  k_init<<<(nNF + BT - 1) / BT, BT, 0, stream>>>(agg, denom, amax, N);

  // 2. pack weights -> WMMA fragment layout + bias vector
  {
    int total = WA_ELEMS + WB_ELEMS + COA;
    k_pack<<<(total + BT - 1) / BT, BT, 0, stream>>>(
        Wq, Wk, Wv, Wskip, Wgate, Wfuse, bq, bk, bv, bskip, WAp, WBp, biasA);
  }

  // 3. nodes -> bf16
  k_f32_to_bf16<<<(nNF + BT - 1) / BT, BT, 0, stream>>>(nodes, nodes_bf, nNF);

  // 4. Y1 = nodes @ [Wq|Wk|Wv|Wskip|Wgate_top] + bias  (WMMA + TDM)
  {
    dim3 grid((N + 127) / 128, COA / 64);
    k_gemm_bf16_wmma<<<grid, BT, 0, stream>>>(nodes_bf, WAp, biasA, Y1, N, COA);
  }

  // 5-7. edge phase: logits+max, exp+denom, weighted scatter-add
  k_edge_logits<<<(E + BT - 1) / BT, BT, 0, stream>>>(srcp, dstp, Y1, alpha, amax, E);
  k_edge_exp<<<(2 * E + BT - 1) / BT, BT, 0, stream>>>(dstp, alpha, amax, denom, E);
  {
    long long threads = (long long)E * 32;
    k_edge_agg<<<(unsigned)((threads + BT - 1) / BT), BT, 0, stream>>>(
        srcp, dstp, Y1, alpha, denom, agg, E);
  }

  // 8. x = agg + skip
  k_make_x<<<(nNF + BT - 1) / BT, BT, 0, stream>>>(agg, Y1, xf, x_bf, N);

  // 9. Y2 = x @ [Wgate_bot|Wfuse]  (WMMA + TDM)
  {
    dim3 grid((N + 127) / 128, COB / 64);
    k_gemm_bf16_wmma<<<grid, BT, 0, stream>>>(x_bf, WBp, nullptr, Y2, N, COB);
  }

  // 10. gated output
  k_output<<<(nNF + BT - 1) / BT, BT, 0, stream>>>(Y1, Y2, nodes, out, N);
}
